// GATNet_58282706206739
// MI455X (gfx1250) — matrix operand order repaired and checked by reading, not compiled, not run on hardware
//
#include <hip/hip_runtime.h>
#include <hip/hip_bf16.h>

#define TPB 256

typedef __attribute__((ext_vector_type(16))) __bf16 v16bf;
typedef __attribute__((ext_vector_type(8)))  float  v8f;

union FragBF { unsigned int u[8]; v16bf bf; };

__device__ __forceinline__ unsigned short f2bf(float f) {
  unsigned u = __float_as_uint(f);
  u += 0x7FFFu + ((u >> 16) & 1u);   // round-to-nearest-even
  return (unsigned short)(u >> 16);
}

// order-preserving float <-> uint map (for atomicMax-based segment max)
__device__ __forceinline__ unsigned enc_ord(float f) {
  unsigned u = __float_as_uint(f);
  return (u & 0x80000000u) ? ~u : (u | 0x80000000u);
}
__device__ __forceinline__ float dec_ord(unsigned u) {
  unsigned b = (u & 0x80000000u) ? (u ^ 0x80000000u) : ~u;
  return __uint_as_float(b);
}

__device__ __forceinline__ float lrelu(float x) { return x > 0.f ? x : 0.2f * x; }

// ---------------------------------------------------------------------------
// C[M,NC] = A[M,K] (fp32, row stride K) @ B[K,NC] (fp32, row stride NC)
// bf16 WMMA, 64x64 output tile per 256-thread block (8 waves, 2 tiles/wave)
// ---------------------------------------------------------------------------
template<int K, int NC>
__global__ __launch_bounds__(256) void gemm_bf16_kernel(
    const float* __restrict__ A, const float* __restrict__ B,
    float* __restrict__ C, int M)
{
  __shared__ unsigned short As[64][34];   // [row][k], +2 pad vs bank conflicts
  __shared__ unsigned short Bs[64][34];   // [n][k]   (B stored transposed)

  const int tid  = threadIdx.x;
  const int wave = tid >> 5;
  const int lane = tid & 31;
  const int ln15 = lane & 15;
  const int ct   = wave & 3;      // column tile 0..3 (n = ct*16..)
  const int rg   = wave >> 1 >> 1;// row group 0..1 (rows rg*16 and rg*16+32)
  const int m0   = blockIdx.x * 64;

  v8f acc0 = {0.f,0.f,0.f,0.f,0.f,0.f,0.f,0.f};
  v8f acc1 = {0.f,0.f,0.f,0.f,0.f,0.f,0.f,0.f};

  const int lr = tid >> 2;        // 0..63 (cooperative-load row / col)
  const int lk = (tid & 3) * 8;   // 0,8,16,24

#pragma unroll
  for (int k0 = 0; k0 < K; k0 += 32) {
    __syncthreads();
    // A tile (64 rows x 32 k), fp32 -> bf16 in LDS
    {
      float va[8];
      if (m0 + lr < M) {
        const float4* p = (const float4*)(A + (size_t)(m0 + lr) * K + k0 + lk);
        float4 f0 = p[0], f1 = p[1];
        va[0]=f0.x; va[1]=f0.y; va[2]=f0.z; va[3]=f0.w;
        va[4]=f1.x; va[5]=f1.y; va[6]=f1.z; va[7]=f1.w;
      } else {
#pragma unroll
        for (int j = 0; j < 8; ++j) va[j] = 0.f;
      }
#pragma unroll
      for (int j = 0; j < 8; ++j) As[lr][lk + j] = f2bf(va[j]);
    }
    // B tile (32 k x 64 n) -> Bs[n][k]
    {
#pragma unroll
      for (int j = 0; j < 8; ++j) {
        int k = lk + j;
        float v = (lr < NC) ? B[(size_t)(k0 + k) * NC + lr] : 0.f;
        Bs[lr][k] = f2bf(v);
      }
    }
    __syncthreads();

    // fragment loads per ISA 16-bit A 16x32 / B 32x16 layouts
    FragBF fb, fa0, fa1;
    const int kbB  = (lane < 16) ? 0 : 16;
    const int kbA  = (lane < 16) ? 0 : 8;
    const int brow = (ct << 4) + ln15;
    const int arow = (rg << 4) + ln15;
#pragma unroll
    for (int v = 0; v < 8; ++v) {
      int ko   = kbA + ((v & 3) << 1) + ((v & 4) << 2);
      fb.u[v]  = *(const unsigned int*)&Bs[brow][ko];
      fa0.u[v] = *(const unsigned int*)&As[arow][ko];
      fa1.u[v] = *(const unsigned int*)&As[arow + 32][ko];
    }
    acc0 = __builtin_amdgcn_wmma_f32_16x16x32_bf16(false, fa0.bf, false, fb.bf,
                                                   (short)0, acc0, false, false);
    acc1 = __builtin_amdgcn_wmma_f32_16x16x32_bf16(false, fa1.bf, false, fb.bf,
                                                   (short)0, acc1, false, false);
  }

  const int nc = (ct << 4) + ln15;
  if (nc < NC) {
    const int mofs = (lane < 16) ? 0 : 8;
#pragma unroll
    for (int r = 0; r < 8; ++r) {
      int m = m0 + (rg << 4) + mofs + r;
      if (m < M)      C[(size_t)m * NC + nc]        = acc0[r];
      if (m + 32 < M) C[(size_t)(m + 32) * NC + nc] = acc1[r];
    }
  }
}

// ---------------------------------------------------------------------------
// Layer-1 attention coefficients: aS[n,h] = <h1[n,h,:], attS[h,:]>, same for aD
// ---------------------------------------------------------------------------
__global__ void alpha1_kernel(const float* __restrict__ h1,
                              const float* __restrict__ attS,
                              const float* __restrict__ attD,
                              float* __restrict__ aS, float* __restrict__ aD, int N)
{
  int g = blockIdx.x * TPB + threadIdx.x;
  if (g >= N * 8) return;
  int n = g >> 3, h = g & 7;
  const float4* hp = (const float4*)(h1 + (size_t)n * 64 + h * 8);
  const float4* sp = (const float4*)(attS + h * 8);
  const float4* dp = (const float4*)(attD + h * 8);
  float4 x0 = hp[0], x1 = hp[1];
  float4 s0 = sp[0], s1 = sp[1];
  float4 d0 = dp[0], d1 = dp[1];
  aS[g] = x0.x*s0.x + x0.y*s0.y + x0.z*s0.z + x0.w*s0.w
        + x1.x*s1.x + x1.y*s1.y + x1.z*s1.z + x1.w*s1.w;
  aD[g] = x0.x*d0.x + x0.y*d0.y + x0.z*d0.z + x0.w*d0.w
        + x1.x*d1.x + x1.y*d1.y + x1.z*d1.z + x1.w*d1.w;
}

__device__ __forceinline__ void edge_sd(const int* __restrict__ ei, int e, int E,
                                        int& s, int& d) {
  if (e < E) { s = ei[e]; d = ei[E + e]; } else { s = d = e - E; }
}

// ---------------------------------------------------------------------------
// Layer-1 segment softmax passes (8 heads per edge, one thread per edge)
// ---------------------------------------------------------------------------
__global__ void edge_max1_kernel(const int* __restrict__ ei, int E, int ET,
                                 const float* __restrict__ aS, const float* __restrict__ aD,
                                 unsigned* __restrict__ mx)
{
  int e = blockIdx.x * TPB + threadIdx.x;
  if (e >= ET) return;
  int s, d; edge_sd(ei, e, E, s, d);
  const float4* ps = (const float4*)(aS + (size_t)s * 8);
  const float4* pd = (const float4*)(aD + (size_t)d * 8);
  float4 s0 = ps[0], s1 = ps[1], d0 = pd[0], d1 = pd[1];
  float sv[8] = {s0.x,s0.y,s0.z,s0.w,s1.x,s1.y,s1.z,s1.w};
  float dv[8] = {d0.x,d0.y,d0.z,d0.w,d1.x,d1.y,d1.z,d1.w};
#pragma unroll
  for (int h = 0; h < 8; ++h)
    atomicMax(&mx[(size_t)d * 8 + h], enc_ord(lrelu(sv[h] + dv[h])));
}

__global__ void edge_sum1_kernel(const int* __restrict__ ei, int E, int ET,
                                 const float* __restrict__ aS, const float* __restrict__ aD,
                                 const unsigned* __restrict__ mx, float* __restrict__ sm)
{
  int e = blockIdx.x * TPB + threadIdx.x;
  if (e >= ET) return;
  int s, d; edge_sd(ei, e, E, s, d);
  const float4* ps = (const float4*)(aS + (size_t)s * 8);
  const float4* pd = (const float4*)(aD + (size_t)d * 8);
  float4 s0 = ps[0], s1 = ps[1], d0 = pd[0], d1 = pd[1];
  float sv[8] = {s0.x,s0.y,s0.z,s0.w,s1.x,s1.y,s1.z,s1.w};
  float dv[8] = {d0.x,d0.y,d0.z,d0.w,d1.x,d1.y,d1.z,d1.w};
#pragma unroll
  for (int h = 0; h < 8; ++h) {
    float m = dec_ord(mx[(size_t)d * 8 + h]);
    atomicAdd(&sm[(size_t)d * 8 + h], __expf(lrelu(sv[h] + dv[h]) - m));
  }
}

// one thread per (edge, head): scatter 8 channels into out1 (L2-resident)
__global__ void agg1_kernel(const int* __restrict__ ei, int E, int ET,
                            const float* __restrict__ aS, const float* __restrict__ aD,
                            const unsigned* __restrict__ mx, const float* __restrict__ sm,
                            const float* __restrict__ h1, float* __restrict__ out1)
{
  int g = blockIdx.x * TPB + threadIdx.x;
  if (g >= ET * 8) return;
  int e = g >> 3, h = g & 7;
  int s, d; edge_sd(ei, e, E, s, d);
  float lg = lrelu(aS[(size_t)s * 8 + h] + aD[(size_t)d * 8 + h]);
  float m  = dec_ord(mx[(size_t)d * 8 + h]);
  float coef = __expf(lg - m) / (sm[(size_t)d * 8 + h] + 1e-16f);
  const float4* hs = (const float4*)(h1 + (size_t)s * 64 + h * 8);
  float4 p = hs[0], q = hs[1];
  float* o = out1 + (size_t)d * 64 + h * 8;
  atomicAdd(o + 0, p.x * coef); atomicAdd(o + 1, p.y * coef);
  atomicAdd(o + 2, p.z * coef); atomicAdd(o + 3, p.w * coef);
  atomicAdd(o + 4, q.x * coef); atomicAdd(o + 5, q.y * coef);
  atomicAdd(o + 6, q.z * coef); atomicAdd(o + 7, q.w * coef);
}

__global__ void bias_elu1_kernel(float* __restrict__ out1, const float* __restrict__ b1, int N)
{
  int g = blockIdx.x * TPB + threadIdx.x;
  if (g >= N * 64) return;
  float v = out1[g] + b1[g & 63];
  out1[g] = v > 0.f ? v : expm1f(v);
}

// ---------------------------------------------------------------------------
// Layer 2 (heads=1, 40 channels)
// ---------------------------------------------------------------------------
__global__ void alpha2_kernel(const float* __restrict__ h2,
                              const float* __restrict__ attS, const float* __restrict__ attD,
                              float* __restrict__ aS, float* __restrict__ aD, int N)
{
  int n = blockIdx.x * TPB + threadIdx.x;
  if (n >= N) return;
  float s = 0.f, d = 0.f;
  const float* hp = h2 + (size_t)n * 40;
#pragma unroll
  for (int c = 0; c < 40; ++c) { float v = hp[c]; s += v * attS[c]; d += v * attD[c]; }
  aS[n] = s; aD[n] = d;
}

__global__ void edge_max2_kernel(const int* __restrict__ ei, int E, int ET,
                                 const float* __restrict__ aS, const float* __restrict__ aD,
                                 unsigned* __restrict__ mx)
{
  int e = blockIdx.x * TPB + threadIdx.x;
  if (e >= ET) return;
  int s, d; edge_sd(ei, e, E, s, d);
  atomicMax(&mx[d], enc_ord(lrelu(aS[s] + aD[d])));
}

__global__ void edge_sum2_kernel(const int* __restrict__ ei, int E, int ET,
                                 const float* __restrict__ aS, const float* __restrict__ aD,
                                 const unsigned* __restrict__ mx, float* __restrict__ sm)
{
  int e = blockIdx.x * TPB + threadIdx.x;
  if (e >= ET) return;
  int s, d; edge_sd(ei, e, E, s, d);
  atomicAdd(&sm[d], __expf(lrelu(aS[s] + aD[d]) - dec_ord(mx[d])));
}

// one thread per (edge, 8-channel chunk); 40 = 5 chunks
__global__ void agg2_kernel(const int* __restrict__ ei, int E, int ET,
                            const float* __restrict__ aS, const float* __restrict__ aD,
                            const unsigned* __restrict__ mx, const float* __restrict__ sm,
                            const float* __restrict__ h2, float* __restrict__ out)
{
  int g = blockIdx.x * TPB + threadIdx.x;
  if (g >= ET * 5) return;
  int e = g / 5, ch = g - e * 5;
  int s, d; edge_sd(ei, e, E, s, d);
  float coef = __expf(lrelu(aS[s] + aD[d]) - dec_ord(mx[d])) / (sm[d] + 1e-16f);
  const float4* hp = (const float4*)(h2 + (size_t)s * 40 + ch * 8);
  float4 p = hp[0], q = hp[1];
  float* o = out + (size_t)d * 40 + ch * 8;
  atomicAdd(o + 0, p.x * coef); atomicAdd(o + 1, p.y * coef);
  atomicAdd(o + 2, p.z * coef); atomicAdd(o + 3, p.w * coef);
  atomicAdd(o + 4, q.x * coef); atomicAdd(o + 5, q.y * coef);
  atomicAdd(o + 6, q.z * coef); atomicAdd(o + 7, q.w * coef);
}

__global__ void bias2_kernel(float* __restrict__ out, const float* __restrict__ b2, int N)
{
  int g = blockIdx.x * TPB + threadIdx.x;
  if (g >= N * 40) return;
  out[g] += b2[g % 40];
}

// ---------------------------------------------------------------------------
extern "C" void kernel_launch(void* const* d_in, const int* in_sizes, int n_in,
                              void* d_out, int out_size, void* d_ws, size_t ws_size,
                              hipStream_t stream)
{
  const float* x     = (const float*)d_in[0];
  const int*   ei    = (const int*)  d_in[1];
  const float* W1    = (const float*)d_in[2];
  const float* attS1 = (const float*)d_in[3];
  const float* attD1 = (const float*)d_in[4];
  const float* b1    = (const float*)d_in[5];
  const float* W2    = (const float*)d_in[6];
  const float* attS2 = (const float*)d_in[7];
  const float* attD2 = (const float*)d_in[8];
  const float* b2    = (const float*)d_in[9];
  float* out = (float*)d_out;

  const int N  = in_sizes[0] / 256;   // F_IN = 256
  const int E  = in_sizes[1] / 2;
  const int ET = E + N;               // with self loops

  // workspace carve (256B aligned segments)
  auto au = [](size_t v) { return (v + 255) & ~size_t(255); };
  char* w = (char*)d_ws;
  float*    h1   = (float*)w;    w += au((size_t)N * 64 * 4);
  float*    out1 = (float*)w;    w += au((size_t)N * 64 * 4);
  float*    aS1  = (float*)w;    w += au((size_t)N * 8 * 4);
  float*    aD1  = (float*)w;    w += au((size_t)N * 8 * 4);
  unsigned* mx1  = (unsigned*)w; w += au((size_t)N * 8 * 4);
  float*    sm1  = (float*)w;    w += au((size_t)N * 8 * 4);
  float*    h2   = (float*)w;    w += au((size_t)N * 40 * 4);
  float*    aS2  = (float*)w;    w += au((size_t)N * 4);
  float*    aD2  = (float*)w;    w += au((size_t)N * 4);
  unsigned* mx2  = (unsigned*)w; w += au((size_t)N * 4);
  float*    sm2  = (float*)w;    w += au((size_t)N * 4);

  auto cdiv = [](int a, int b) { return (a + b - 1) / b; };

  // ---- layer 1 ----
  gemm_bf16_kernel<256, 64><<<cdiv(N, 64), 256, 0, stream>>>(x, W1, h1, N);
  alpha1_kernel<<<cdiv(N * 8, TPB), TPB, 0, stream>>>(h1, attS1, attD1, aS1, aD1, N);
  hipMemsetAsync(mx1, 0, (size_t)N * 8 * 4, stream);   // 0 < encoded(-inf): floor below all values
  hipMemsetAsync(sm1, 0, (size_t)N * 8 * 4, stream);
  hipMemsetAsync(out1, 0, (size_t)N * 64 * 4, stream);
  edge_max1_kernel<<<cdiv(ET, TPB), TPB, 0, stream>>>(ei, E, ET, aS1, aD1, mx1);
  edge_sum1_kernel<<<cdiv(ET, TPB), TPB, 0, stream>>>(ei, E, ET, aS1, aD1, mx1, sm1);
  agg1_kernel<<<cdiv(ET * 8, TPB), TPB, 0, stream>>>(ei, E, ET, aS1, aD1, mx1, sm1, h1, out1);
  bias_elu1_kernel<<<cdiv(N * 64, TPB), TPB, 0, stream>>>(out1, b1, N);

  // ---- layer 2 ----
  gemm_bf16_kernel<64, 40><<<cdiv(N, 64), 256, 0, stream>>>(out1, W2, h2, N);
  alpha2_kernel<<<cdiv(N, TPB), TPB, 0, stream>>>(h2, attS2, attD2, aS2, aD2, N);
  hipMemsetAsync(mx2, 0, (size_t)N * 4, stream);
  hipMemsetAsync(sm2, 0, (size_t)N * 4, stream);
  hipMemsetAsync(out, 0, (size_t)N * 40 * 4, stream);
  edge_max2_kernel<<<cdiv(ET, TPB), TPB, 0, stream>>>(ei, E, ET, aS2, aD2, mx2);
  edge_sum2_kernel<<<cdiv(ET, TPB), TPB, 0, stream>>>(ei, E, ET, aS2, aD2, mx2, sm2);
  agg2_kernel<<<cdiv(ET * 5, TPB), TPB, 0, stream>>>(ei, E, ET, aS2, aD2, mx2, sm2, h2, out);
  bias2_kernel<<<cdiv(N * 40, TPB), TPB, 0, stream>>>(out, b2, N);
}
